// TransformerBlock_54692113547943
// MI455X (gfx1250) — compile-verified, matrix operand order checked by reading, not run on hardware
//
#include <hip/hip_runtime.h>
#include <math.h>

// ---------------------------------------------------------------------------
// Types for CDNA5 WMMA (wave32): V_WMMA_F32_16X16X32_BF16
// ---------------------------------------------------------------------------
typedef __bf16 bf16_t;
typedef __attribute__((ext_vector_type(8)))  __bf16 bf16x8;
typedef __attribute__((ext_vector_type(16))) __bf16 bf16x16;
typedef __attribute__((ext_vector_type(8)))  float  floatx8;

#define HIDDEN 1024
#define TOKENS 4096   // B*T
#define SEQ    2048
#define NH     16
#define DK     64
#define FFN    4096
#define C3     3072

__device__ __forceinline__ floatx8 fzero() {
    floatx8 z;
#pragma unroll
    for (int i = 0; i < 8; ++i) z[i] = 0.0f;
    return z;
}

__device__ __forceinline__ floatx8 wmma_bf16(bf16x16 a, bf16x16 b, floatx8 c) {
    // D = A(16x32) * B(32x16) + C, f32 accumulate
    return __builtin_amdgcn_wmma_f32_16x16x32_bf16(
        false, a, false, b, (short)0, c, false, false);
}

// Load one 16x32 bf16 fragment (A layout; B uses the same per-lane pattern on
// W^T / K / V^T rows). ISA 7.12.2: lane L holds row (L&15); lanes 0-15 hold
// K = 0..7 and 16..23, lanes 16-31 hold K = 8..15 and 24..31.
__device__ __forceinline__ bf16x16 load_frag(const bf16_t* base, int stride, int lane) {
    int r = lane & 15;
    int h = (lane >> 4) & 1;
    const bf16_t* p = base + (size_t)r * stride + h * 8;
    union { bf16x16 v; bf16x8 q[2]; } u;
    u.q[0] = *(const bf16x8*)(p);
    u.q[1] = *(const bf16x8*)(p + 16);
    return u.v;
}

__device__ __forceinline__ float wave_red_add(float v) {
#pragma unroll
    for (int off = 16; off > 0; off >>= 1) v += __shfl_xor(v, off, 32);
    return v;
}

// ---------------------------------------------------------------------------
// Weight transpose + cast:  out[c][r] = (bf16) in[r][c]   (in: R x C fp32)
// ---------------------------------------------------------------------------
__global__ void transpose_cast_kernel(const float* __restrict__ in,
                                      bf16_t* __restrict__ out, int R, int C) {
    __shared__ float tile[32][33];
    int c0 = blockIdx.x * 32, r0 = blockIdx.y * 32;
    int tx = threadIdx.x, ty = threadIdx.y;
#pragma unroll
    for (int i = 0; i < 32; i += 8)
        tile[ty + i][tx] = in[(size_t)(r0 + ty + i) * C + c0 + tx];
    __syncthreads();
#pragma unroll
    for (int i = 0; i < 32; i += 8)
        out[(size_t)(c0 + ty + i) * R + r0 + tx] = (bf16_t)tile[tx][ty + i];
}

// ---------------------------------------------------------------------------
// LayerNorm over last dim (1024) -> bf16. One block (256 thr) per token row.
// ---------------------------------------------------------------------------
__global__ __launch_bounds__(256) void ln_kernel(const float* __restrict__ x,
                                                 const float* __restrict__ g,
                                                 const float* __restrict__ be,
                                                 bf16_t* __restrict__ out) {
    int row = blockIdx.x;
    const float4 f4 = ((const float4*)(x + (size_t)row * HIDDEN))[threadIdx.x];
    int lane = threadIdx.x & 31, wv = threadIdx.x >> 5;
    __shared__ float part[8];

    float s = f4.x + f4.y + f4.z + f4.w;
    s = wave_red_add(s);
    if (lane == 0) part[wv] = s;
    __syncthreads();
    float tot = 0.0f;
#pragma unroll
    for (int i = 0; i < 8; ++i) tot += part[i];
    float mu = tot * (1.0f / HIDDEN);

    float d0 = f4.x - mu, d1 = f4.y - mu, d2 = f4.z - mu, d3 = f4.w - mu;
    float q = d0 * d0 + d1 * d1 + d2 * d2 + d3 * d3;
    q = wave_red_add(q);
    __syncthreads();
    if (lane == 0) part[wv] = q;
    __syncthreads();
    float vt = 0.0f;
#pragma unroll
    for (int i = 0; i < 8; ++i) vt += part[i];
    float rstd = rsqrtf(vt * (1.0f / HIDDEN) + 1e-5f);

    int c = threadIdx.x * 4;
    bf16_t* op = out + (size_t)row * HIDDEN + c;
    op[0] = (bf16_t)(d0 * rstd * g[c + 0] + be[c + 0]);
    op[1] = (bf16_t)(d1 * rstd * g[c + 1] + be[c + 1]);
    op[2] = (bf16_t)(d2 * rstd * g[c + 2] + be[c + 2]);
    op[3] = (bf16_t)(d3 * rstd * g[c + 3] + be[c + 3]);
}

// ---------------------------------------------------------------------------
// Repack V into V^T per head:  vT[(b*16+h)][d][t] = qkv[b*T+t][2048+h*64+d]
// ---------------------------------------------------------------------------
__global__ void vtrans_kernel(const bf16_t* __restrict__ qkv,
                              bf16_t* __restrict__ vT) {
    int bh = blockIdx.z; int b = bh >> 4, h = bh & 15;
    __shared__ bf16_t tile[32][33];
    int d0 = blockIdx.x * 32, t0 = blockIdx.y * 32;
    int tx = threadIdx.x, ty = threadIdx.y;
#pragma unroll
    for (int i = 0; i < 32; i += 8)
        tile[ty + i][tx] =
            qkv[(size_t)(b * SEQ + t0 + ty + i) * C3 + 2048 + h * DK + d0 + tx];
    __syncthreads();
#pragma unroll
    for (int i = 0; i < 32; i += 8)
        vT[(size_t)bh * DK * SEQ + (size_t)(d0 + ty + i) * SEQ + t0 + tx] =
            tile[tx][ty + i];
}

// ---------------------------------------------------------------------------
// WMMA GEMM: C[M,N] = A[M,K](bf16) * Bt[N,K]^T(bf16) + bias, epilogues:
//   EPI=0: store bf16 (QKV)         EPI=1: + res, store f32 (residual adds)
//   EPI=2: exact GELU, store bf16 (FF1)
// Block: 256 thr (8 waves). Tile 128x128, BK=32. Wave = 32(M) x 64(N).
// Double-buffered LDS (2 x 20 KB): next tile's global loads are issued into
// registers, the current tile's WMMAs run, then the staged registers are
// written to the other LDS buffer -> one barrier per K step.
// ---------------------------------------------------------------------------
template <int EPI>
__global__ __launch_bounds__(256) void gemm_kernel(
    const bf16_t* __restrict__ A, const bf16_t* __restrict__ Bt,
    const float* __restrict__ bias, const float* __restrict__ res,
    void* __restrict__ outv, int M, int N, int K) {
    // Row stride 40 elems = 80B (16B multiple) -> aligned b128 LDS accesses.
    __shared__ bf16_t As[2][128][40];
    __shared__ bf16_t Bs[2][128][40];

    int tid = threadIdx.x;
    int lane = tid & 31, wv = tid >> 5;
    int wm = wv & 3, wn = wv >> 2;           // 4 waves along M, 2 along N
    size_t m0 = (size_t)blockIdx.y * 128;
    size_t n0 = (size_t)blockIdx.x * 128;

    floatx8 acc[2][4];
#pragma unroll
    for (int i = 0; i < 2; ++i)
#pragma unroll
        for (int j = 0; j < 4; ++j) acc[i][j] = fzero();

    int r  = tid >> 2;                        // 0..63
    int cg = (tid & 3) * 8;                   // 0,8,16,24
    int KT = K >> 5;                          // K/32 steps

    // Prologue: stage tile 0 into buffer 0.
#pragma unroll
    for (int p = 0; p < 2; ++p) {
        int row = r + p * 64;
        *(bf16x8*)&As[0][row][cg] = *(const bf16x8*)&A[(m0 + row) * (size_t)K + cg];
        *(bf16x8*)&Bs[0][row][cg] = *(const bf16x8*)&Bt[(n0 + row) * (size_t)K + cg];
    }
    __syncthreads();

    for (int kt = 0; kt < KT; ++kt) {
        int buf = kt & 1;
        bool haveNext = (kt + 1 < KT);
        bf16x8 ra[2], rb[2];
        if (haveNext) {
            int k0 = (kt + 1) * 32;
#pragma unroll
            for (int p = 0; p < 2; ++p) {
                int row = r + p * 64;
                ra[p] = *(const bf16x8*)&A[(m0 + row) * (size_t)K + k0 + cg];
                rb[p] = *(const bf16x8*)&Bt[(n0 + row) * (size_t)K + k0 + cg];
            }
            if (kt + 2 < KT) {
                int k2 = (kt + 2) * 32;
                // gfx1250 global_prefetch_b8 into WGP caches (L2-resident data)
                __builtin_prefetch(&A[(m0 + r) * (size_t)K + k2 + cg], 0, 3);
                __builtin_prefetch(&Bt[(n0 + r) * (size_t)K + k2 + cg], 0, 3);
            }
        }

        bf16x16 af[2], bfr[4];
#pragma unroll
        for (int i = 0; i < 2; ++i)
            af[i] = load_frag(&As[buf][wm * 32 + i * 16][0], 40, lane);
#pragma unroll
        for (int j = 0; j < 4; ++j)
            bfr[j] = load_frag(&Bs[buf][wn * 64 + j * 16][0], 40, lane);
#pragma unroll
        for (int i = 0; i < 2; ++i)
#pragma unroll
            for (int j = 0; j < 4; ++j)
                acc[i][j] = wmma_bf16(af[i], bfr[j], acc[i][j]);

        if (haveNext) {
            int nb = buf ^ 1;
#pragma unroll
            for (int p = 0; p < 2; ++p) {
                int row = r + p * 64;
                *(bf16x8*)&As[nb][row][cg] = ra[p];
                *(bf16x8*)&Bs[nb][row][cg] = rb[p];
            }
        }
        __syncthreads();
    }

    // Epilogue. D layout: lane holds column (lane&15); rows 8*(lane>>4)+e.
    int cl = lane & 15, hf = lane >> 4;
#pragma unroll
    for (int i = 0; i < 2; ++i)
#pragma unroll
        for (int j = 0; j < 4; ++j) {
            size_t col = n0 + wn * 64 + j * 16 + cl;
            float bcol = bias[col];
#pragma unroll
            for (int e = 0; e < 8; ++e) {
                size_t row = m0 + wm * 32 + i * 16 + hf * 8 + e;
                float v = acc[i][j][e] + bcol;
                if constexpr (EPI == 0) {
                    ((bf16_t*)outv)[row * N + col] = (bf16_t)v;
                } else if constexpr (EPI == 1) {
                    ((float*)outv)[row * N + col] = v + res[row * N + col];
                } else {
                    float gl = 0.5f * v * (1.0f + erff(v * 0.70710678118654752f));
                    ((bf16_t*)outv)[row * N + col] = (bf16_t)gl;
                }
            }
        }
}

// ---------------------------------------------------------------------------
// Flash attention. 256 thr = 8 waves; each wave owns one 16-row Q tile of one
// (b,h). K-chunks of 64 columns: QK^T via WMMA, online softmax with 16-lane
// shuffles (D-layout row groups), P transposed through per-wave LDS into
// A-fragment layout, then P*V WMMAs against V^T.
// ---------------------------------------------------------------------------
__global__ __launch_bounds__(256) void attn_kernel(
    const bf16_t* __restrict__ qkv, const bf16_t* __restrict__ vT,
    const int* __restrict__ mask, bf16_t* __restrict__ ctx) {
    // per-wave 16x64 P tile, row stride 80 elems = 160B (16B multiple)
    __shared__ bf16_t Pl[8][16][80];

    int lane = threadIdx.x & 31, wv = threadIdx.x >> 5;
    int gw = blockIdx.x * 8 + wv;
    int qt = gw & 127;               // 128 q-tiles of 16 rows
    int h  = (gw >> 7) & 15;
    int b  = gw >> 11;
    int cl = lane & 15, hf = lane >> 4;

    const bf16_t* qbase =
        qkv + (size_t)(b * SEQ + qt * 16) * C3 + (size_t)h * DK;
    bf16x16 qa0 = load_frag(qbase,      C3, lane);   // d = 0..31
    bf16x16 qa1 = load_frag(qbase + 32, C3, lane);   // d = 32..63

    floatx8 acc[4];
#pragma unroll
    for (int j = 0; j < 4; ++j) acc[j] = fzero();
    float m[8], l[8];
#pragma unroll
    for (int e = 0; e < 8; ++e) { m[e] = -INFINITY; l[e] = 0.0f; }

    const bf16_t* vhead = vT + (size_t)(b * NH + h) * DK * SEQ;

    for (int t0 = 0; t0 < SEQ; t0 += 64) {
        // Prefetch next chunk's K rows (one lane -> one row's cacheline).
        if (t0 + 64 < SEQ) {
            __builtin_prefetch(
                qkv + (size_t)(b * SEQ + t0 + 64 + lane) * C3 + 1024 + (size_t)h * DK,
                0, 3);
        }
        // ---- S = Q K^T for 4 sub-tiles of 16 key columns -------------------
        floatx8 S[4];
#pragma unroll
        for (int sub = 0; sub < 4; ++sub) {
            const bf16_t* kb =
                qkv + (size_t)(b * SEQ + t0 + sub * 16) * C3 + 1024 + (size_t)h * DK;
            bf16x16 k0 = load_frag(kb,      C3, lane);
            bf16x16 k1 = load_frag(kb + 32, C3, lane);
            floatx8 z = fzero();
            z = wmma_bf16(qa0, k0, z);
            z = wmma_bf16(qa1, k1, z);
            S[sub] = z;
        }
        // ---- scale + mask (lane's column is fixed -> scalar mask) ----------
#pragma unroll
        for (int sub = 0; sub < 4; ++sub) {
            int col = t0 + sub * 16 + cl;
            bool mz = (mask[b * SEQ + col] == 0);
#pragma unroll
            for (int e = 0; e < 8; ++e) {
                float s = S[sub][e] * 0.125f;   // 1/sqrt(DK)
                S[sub][e] = mz ? -INFINITY : s;
            }
        }
        // ---- online softmax: row max over 4 tiles + 16-lane group ----------
        float tmax[8];
#pragma unroll
        for (int e = 0; e < 8; ++e)
            tmax[e] = fmaxf(fmaxf(S[0][e], S[1][e]), fmaxf(S[2][e], S[3][e]));
#pragma unroll
        for (int off = 1; off < 16; off <<= 1)
#pragma unroll
            for (int e = 0; e < 8; ++e)
                tmax[e] = fmaxf(tmax[e], __shfl_xor(tmax[e], off, 32));
        float alpha[8];
#pragma unroll
        for (int e = 0; e < 8; ++e) {
            float mn = fmaxf(m[e], tmax[e]);
            alpha[e] = (m[e] == -INFINITY) ? 0.0f : __expf(m[e] - mn);
            m[e] = mn;
        }
        // ---- P = exp(S-m), row sums, stash P (bf16) to LDS for transpose ---
        float rsum[8];
#pragma unroll
        for (int e = 0; e < 8; ++e) rsum[e] = 0.0f;
#pragma unroll
        for (int sub = 0; sub < 4; ++sub)
#pragma unroll
            for (int e = 0; e < 8; ++e) {
                float s = S[sub][e];
                float p = (s == -INFINITY) ? 0.0f : __expf(s - m[e]);
                rsum[e] += p;
                Pl[wv][hf * 8 + e][sub * 16 + cl] = (bf16_t)p;
            }
#pragma unroll
        for (int off = 1; off < 16; off <<= 1)
#pragma unroll
            for (int e = 0; e < 8; ++e) rsum[e] += __shfl_xor(rsum[e], off, 32);
#pragma unroll
        for (int e = 0; e < 8; ++e) l[e] = l[e] * alpha[e] + rsum[e];
#pragma unroll
        for (int j = 0; j < 4; ++j)
#pragma unroll
            for (int e = 0; e < 8; ++e) acc[j][e] *= alpha[e];

        __syncthreads();   // uniform trip count -> legal; orders LDS P tile
        bf16x16 pa0 = load_frag(&Pl[wv][0][0],  80, lane);  // k-cols 0..31
        bf16x16 pa1 = load_frag(&Pl[wv][0][32], 80, lane);  // k-cols 32..63

        // ---- ctx += P * V  (V^T rows are contiguous along t) ---------------
#pragma unroll
        for (int j = 0; j < 4; ++j) {
            const bf16_t* vb = vhead + (size_t)(j * 16) * SEQ + t0;
            bf16x16 v0 = load_frag(vb,      SEQ, lane);
            bf16x16 v1 = load_frag(vb + 32, SEQ, lane);
            acc[j] = wmma_bf16(pa0, v0, acc[j]);
            acc[j] = wmma_bf16(pa1, v1, acc[j]);
        }
        __syncthreads();
    }

    // ---- normalize and store ctx (bf16) ------------------------------------
#pragma unroll
    for (int j = 0; j < 4; ++j)
#pragma unroll
        for (int e = 0; e < 8; ++e) {
            size_t row = (size_t)b * SEQ + qt * 16 + hf * 8 + e;
            float o = (l[e] > 0.0f) ? acc[j][e] / l[e] : 0.0f;
            ctx[row * HIDDEN + h * DK + j * 16 + cl] = (bf16_t)o;
        }
}

// ---------------------------------------------------------------------------
// Launcher
// ---------------------------------------------------------------------------
extern "C" void kernel_launch(void* const* d_in, const int* in_sizes, int n_in,
                              void* d_out, int out_size, void* d_ws, size_t ws_size,
                              hipStream_t stream) {
    const float* x      = (const float*)d_in[0];
    const int*   mask   = (const int*)  d_in[1];
    const float* W_qkv  = (const float*)d_in[2];
    const float* b_qkv  = (const float*)d_in[3];
    const float* W_out  = (const float*)d_in[4];
    const float* b_out  = (const float*)d_in[5];
    const float* W_ff1  = (const float*)d_in[6];
    const float* b_ff1  = (const float*)d_in[7];
    const float* W_ff2  = (const float*)d_in[8];
    const float* b_ff2  = (const float*)d_in[9];
    const float* g_ln1  = (const float*)d_in[10];
    const float* be_ln1 = (const float*)d_in[11];
    const float* g_ln2  = (const float*)d_in[12];
    const float* be_ln2 = (const float*)d_in[13];

    // Workspace layout (~92 MB). act reuses the qkv+vT span (exactly 32 MB).
    char* w = (char*)d_ws;
    bf16_t* Wqkv_t = (bf16_t*)w; w += (size_t)C3 * HIDDEN * 2;
    bf16_t* Wout_t = (bf16_t*)w; w += (size_t)HIDDEN * HIDDEN * 2;
    bf16_t* Wff1_t = (bf16_t*)w; w += (size_t)FFN * HIDDEN * 2;
    bf16_t* Wff2_t = (bf16_t*)w; w += (size_t)HIDDEN * FFN * 2;
    bf16_t* hbuf   = (bf16_t*)w; w += (size_t)TOKENS * HIDDEN * 2;
    bf16_t* qkv    = (bf16_t*)w; bf16_t* act = qkv;
                                 w += (size_t)TOKENS * C3 * 2;
    bf16_t* vT     = (bf16_t*)w; w += (size_t)2 * NH * DK * SEQ * 2;
    bf16_t* ctx    = (bf16_t*)w; w += (size_t)TOKENS * HIDDEN * 2;
    float*  x1     = (float*)w;  w += (size_t)TOKENS * HIDDEN * 4;

    dim3 tb(32, 8);
    // Weight transpose+cast to bf16 W^T
    transpose_cast_kernel<<<dim3(C3 / 32, HIDDEN / 32), tb, 0, stream>>>(W_qkv, Wqkv_t, HIDDEN, C3);
    transpose_cast_kernel<<<dim3(HIDDEN / 32, HIDDEN / 32), tb, 0, stream>>>(W_out, Wout_t, HIDDEN, HIDDEN);
    transpose_cast_kernel<<<dim3(FFN / 32, HIDDEN / 32), tb, 0, stream>>>(W_ff1, Wff1_t, HIDDEN, FFN);
    transpose_cast_kernel<<<dim3(HIDDEN / 32, FFN / 32), tb, 0, stream>>>(W_ff2, Wff2_t, FFN, HIDDEN);

    // LN1 -> h
    ln_kernel<<<TOKENS, 256, 0, stream>>>(x, g_ln1, be_ln1, hbuf);
    // qkv = h @ W_qkv + b_qkv (bf16)
    gemm_kernel<0><<<dim3(C3 / 128, TOKENS / 128), 256, 0, stream>>>(
        hbuf, Wqkv_t, b_qkv, nullptr, qkv, TOKENS, C3, HIDDEN);
    // V^T repack
    vtrans_kernel<<<dim3(DK / 32, SEQ / 32, 2 * NH), tb, 0, stream>>>(qkv, vT);
    // attention -> ctx
    attn_kernel<<<512, 256, 0, stream>>>(qkv, vT, mask, ctx);
    // x1 = x + ctx @ W_out + b_out (f32)
    gemm_kernel<1><<<dim3(HIDDEN / 128, TOKENS / 128), 256, 0, stream>>>(
        ctx, Wout_t, b_out, x, x1, TOKENS, HIDDEN, HIDDEN);
    // LN2 -> h
    ln_kernel<<<TOKENS, 256, 0, stream>>>(x1, g_ln2, be_ln2, hbuf);
    // act = gelu(h @ W_ff1 + b_ff1) (bf16)
    gemm_kernel<2><<<dim3(FFN / 128, TOKENS / 128), 256, 0, stream>>>(
        hbuf, Wff1_t, b_ff1, nullptr, act, TOKENS, FFN, HIDDEN);
    // out = x1 + act @ W_ff2 + b_ff2 (f32)
    gemm_kernel<1><<<dim3(HIDDEN / 128, TOKENS / 128), 256, 0, stream>>>(
        act, Wff2_t, b_ff2, x1, (float*)d_out, TOKENS, HIDDEN, FFN);
}
